// ScalarLSTM_36352603194015
// MI455X (gfx1250) — matrix operand order repaired and checked by reading, not compiled, not run on hardware
//
#include <hip/hip_runtime.h>

#define Bsz 128
#define Ssz 512
#define Isz 1024
#define Hsz 1024
#define Osz 1024
#define KK  (Isz + Hsz)      // 2048
#define BH  (Bsz * Hsz)      // 131072

#define KC    128            // staged K-chunk (elems) = 256B per row
#define BROW  (KC + 8)       // LDS row stride (elems): 136 -> conflict-free frag reads
#define GROW  65             // LDS G-exchange row stride (floats)

typedef __bf16 bf16;
typedef bf16  v16bf __attribute__((ext_vector_type(16)));
typedef bf16  v8bf  __attribute__((ext_vector_type(8)));
typedef float v8f   __attribute__((ext_vector_type(8)));
typedef unsigned int v4u __attribute__((ext_vector_type(4)));
typedef int          v4i __attribute__((ext_vector_type(4)));
typedef int          v8i __attribute__((ext_vector_type(8)));

// float -> bf16 RNE without relying on __bf16 arithmetic
static __device__ __forceinline__ bf16 f2bf(float f) {
  union { float f; unsigned u; } v; v.f = f;
  unsigned r = v.u + 0x7FFFu + ((v.u >> 16) & 1u);
  unsigned short hs = (unsigned short)(r >> 16);
  return __builtin_bit_cast(bf16, hs);
}

static __device__ __forceinline__ v16bf cat8(v8bf lo, v8bf hi) {
  return __builtin_shufflevector(lo, hi, 0,1,2,3,4,5,6,7,8,9,10,11,12,13,14,15);
}

// ---- TDM: DMA a (padded) weight slab global -> LDS --------------------------
// D# per CDNA5 ISA.  data_size = 8B units.
//   x: tile0u units of 8B, contiguous               (one 256B row: tile0u=32)
//   y: tile1 rows, stride s0u (8B units)
//   z: tile2 slabs, stride s1u (8B units; 0/unused for 2D)
// pad_enable with pad_interval=5 (64 DWORDs = 256B) and pad_amount=3 (4 DWORDs
// = 16B) inserts exactly the +8-element row padding the frag readers expect.
// This toolchain exposes the 6-arg builtin; the trailing int32x8 slot is
// zero-filled (probe-verified form).
static __device__ __forceinline__ void tdm_load_to_lds(
    const void* gsrc, unsigned ldsOff, unsigned tile0u, unsigned tile1,
    unsigned tile2, unsigned long long s0u, unsigned long long s1u) {
  const unsigned long long ga = (unsigned long long)gsrc;
  v4u g0;
  g0[0] = 1u;                                              // count=1 (valid user D#)
  g0[1] = ldsOff;                                          // LDS byte address
  g0[2] = (unsigned)ga;                                    // global_addr[31:0]
  g0[3] = (unsigned)((ga >> 32) & 0x01FFFFFFu) | (2u << 30); // addr[56:32] | type=2
  // group1 (256b) assembled as 4 qwords:
  const unsigned long long q0 =                            // bits [63:0]
      (3ull << 16)                                         // data_size = 8B
    | (1ull << 20)                                         // pad_enable
    | (5ull << 22)                                         // pad_interval = 64 DW
    | (3ull << 25);                                        // pad_amount = 4 DW
  const unsigned long long q1 =                            // bits [127:64]
      0x4000ull                                            // tensor_dim0 = 2^30 (hi16)
    | (0x40000000ull << 16)                                // tensor_dim1 = 2^30
    | ((unsigned long long)tile0u << 48);                  // tile_dim0
  const unsigned long long q2 =                            // bits [191:128]
      (unsigned long long)tile1                            // tile_dim1
    | ((unsigned long long)tile2 << 16)                    // tile_dim2 (0 = 2D)
    | ((s0u & 0xFFFFFFFFull) << 32);                       // dim0 stride lo
  const unsigned long long q3 =                            // bits [255:192]
      (s0u >> 32)                                          // dim0 stride hi
    | (s1u << 16);                                         // dim1 stride (48b)
  v8i g1 = { (int)(unsigned)q0, (int)(q0 >> 32),
             (int)(unsigned)q1, (int)(q1 >> 32),
             (int)(unsigned)q2, (int)(q2 >> 32),
             (int)(unsigned)q3, (int)(q3 >> 32) };
  v4i g2 = { 0x40000000, 1, 0, 0 };   // tensor_dim2 big; dim3/stride2 unused
  v4i g3 = { 0, 0, 0, 0 };
  v8i g4 = { 0, 0, 0, 0, 0, 0, 0, 0 };                     // unused slot (zero)
  __builtin_amdgcn_tensor_load_to_lds(g0, g1, g2, g3, g4, 0);
}

// ---- WMMA fragment loaders --------------------------------------------------
// A (16x32, MxK): lane holds row m0+(lane&15), chunks K = kTile+(lane>>4)*8..+7
// and +16..+23 (ISA 16-bit A table). comb = [x_t ; h]; K-tiles are 32-aligned
// and I is a multiple of 32, so a tile never straddles the x/h boundary.
static __device__ __forceinline__ v16bf load_A_comb(
    const bf16* __restrict__ xb, const bf16* __restrict__ hb,
    int m0, int t, int kTile, int lane) {
  const int row = m0 + (lane & 15);
  const int k0  = kTile + (lane >> 4) * 8;
  const bf16* src = (kTile < Isz)
      ? (xb + ((size_t)row * Ssz + t) * Isz)
      : (hb + (size_t)row * Hsz - Isz);        // src[k] == hb[row*H + k - I]
  v8bf lo = *(const v8bf*)(src + k0);
  v8bf hi = *(const v8bf*)(src + k0 + 16);
  return cat8(lo, hi);
}

static __device__ __forceinline__ v16bf load_A_h(
    const bf16* __restrict__ hb, int m0, int kTile, int lane) {
  const int row = m0 + (lane & 15);
  const int k0  = kTile + (lane >> 4) * 8;
  const bf16* src = hb + (size_t)row * Hsz;
  v8bf lo = *(const v8bf*)(src + k0);
  v8bf hi = *(const v8bf*)(src + k0 + 16);
  return cat8(lo, hi);
}

// B (32x16, KxN) fragment from LDS-staged N-major slab (row stride BROW elems):
// lane holds N = tile+(lane&15), K = kk+(lane>>4)*8 .. +7 and +16..+23 (16B-aligned pieces)
static __device__ __forceinline__ v16bf lds_fragB(const bf16* p) {
  v8bf lo = *(const v8bf*)(p);
  v8bf hi = *(const v8bf*)(p + 16);
  return cat8(lo, hi);
}

#define WMMA_BF16(acc, A, B) \
  acc = __builtin_amdgcn_wmma_f32_16x16x32_bf16(false, (A), false, (B), (short)0, (acc), false, false)

// ---- Fused per-step kernel --------------------------------------------------
// grid = 16 blocks (one per 64 output columns), block = 1024 thr = 32 waves.
// Wave w: gate = w>>3, M-rows 16*(w&7)..  TDM (wave 0) double-buffers the
// 4-gate weight slab into LDS while all waves run WMMA; accumulators are
// exchanged through LDS and the stabilized-LSTM update runs in-block.
__global__ void __launch_bounds__(1024)
lstm_step_fused(const bf16* __restrict__ xb, const bf16* __restrict__ hbPrev,
                const bf16* __restrict__ WT,      // [4][Hsz][KK] N-major
                const float* __restrict__ bf_, const float* __restrict__ bi_,
                const float* __restrict__ bc_, const float* __restrict__ bo_,
                float* __restrict__ h, float* __restrict__ c,
                float* __restrict__ n, float* __restrict__ m,
                bf16* __restrict__ hbNew, int t) {
  extern __shared__ char smem[];
  bf16* ldsB = (bf16*)smem;                 // 2 buffers x (4*64 rows) x BROW
  const size_t BUFE = (size_t)4 * 64 * BROW;           // elems per buffer
  const unsigned BUFBYTES = (unsigned)(BUFE * 2);      // 69,632 B

  const int tid  = threadIdx.x;
  const int lane = tid & 31;
  const int wave = tid >> 5;                // 0..31
  const int gate = wave >> 3;               // 0..3
  const int m0   = (wave & 7) * 16;
  const int nBlock = blockIdx.x * 64;
  const unsigned ldsBase = __builtin_amdgcn_groupstaticsize();
  const bf16* wSlab = WT + (size_t)nBlock * KK;        // gate-0 row of this block

  const int NST = KK / KC;                  // 16 stages
  // prologue: TDM stage 0 into buffer 0 (3D tile: 256B x 64 rows x 4 gates)
  if (wave == 0) {
    tdm_load_to_lds(wSlab, ldsBase, 32u, 64u, 4u,
                    (unsigned long long)(KK * 2 / 8),
                    (unsigned long long)Hsz * KK * 2 / 8);
    __builtin_amdgcn_s_wait_tensorcnt(0);
  }
  __syncthreads();

  v8f a0 = {}, a1 = {}, a2 = {}, a3 = {};
  const bf16* slab0 = ldsB + (size_t)gate * 64 * BROW;
  const int fn = lane & 15;
  const int fko = (lane >> 4) * 8;

  for (int s = 0; s < NST; ++s) {
    const bf16* cur = slab0 + (size_t)(s & 1) * BUFE;
    const bool more = (s + 1 < NST);
    if (more && wave == 0) {                // DMA next stage while computing
      tdm_load_to_lds(wSlab + (size_t)(s + 1) * KC,
                      ldsBase + (unsigned)((s + 1) & 1) * BUFBYTES,
                      32u, 64u, 4u,
                      (unsigned long long)(KK * 2 / 8),
                      (unsigned long long)Hsz * KK * 2 / 8);
    }
#pragma unroll
    for (int kk = 0; kk < KC; kk += 32) {
      v16bf A = load_A_comb(xb, hbPrev, m0, t, s * KC + kk, lane);
      const bf16* fb = cur + (size_t)fn * BROW + kk + fko;
      v16bf B0 = lds_fragB(fb);
      v16bf B1 = lds_fragB(fb + 16 * BROW);
      v16bf B2 = lds_fragB(fb + 32 * BROW);
      v16bf B3 = lds_fragB(fb + 48 * BROW);
      WMMA_BF16(a0, A, B0);
      WMMA_BF16(a1, A, B1);
      WMMA_BF16(a2, A, B2);
      WMMA_BF16(a3, A, B3);
    }
    if (more) {
      if (wave == 0) __builtin_amdgcn_s_wait_tensorcnt(0);
      __syncthreads();                      // TDM data visible to all waves
    }
  }

  // exchange accumulators through LDS (reuse staging buffer)
  __syncthreads();
  float* ldsG = (float*)smem;               // [4][128][GROW]
  const int half = lane >> 4;
#pragma unroll
  for (int r = 0; r < 8; ++r) {
    float* gp = ldsG + ((size_t)gate * Bsz + (m0 + r + 8 * half)) * GROW + fn;
    gp[ 0] = a0[r];
    gp[16] = a1[r];
    gp[32] = a2[r];
    gp[48] = a3[r];
  }
  __syncthreads();

  // stabilized-LSTM pointwise update for this block's 128 x 64 elements
  for (int e = tid; e < Bsz * 64; e += 1024) {
    const int b   = e >> 6;
    const int nn  = e & 63;
    const int col = nBlock + nn;
    const int idx = b * Hsz + col;
    const float fLog = ldsG[((size_t)0 * Bsz + b) * GROW + nn] + bf_[col];
    const float iLog = ldsG[((size_t)1 * Bsz + b) * GROW + nn] + bi_[col];
    const float cIn  = ldsG[((size_t)2 * Bsz + b) * GROW + nn] + bc_[col];
    const float oIn  = ldsG[((size_t)3 * Bsz + b) * GROW + nn] + bo_[col];
    const float mo = m[idx];
    const float mn = fmaxf(fLog + mo, iLog);
    const float it = __expf(iLog - mn);
    const float ft = __expf(fLog + mo - mn);
    const float ch = tanhf(cIn);
    const float ot = 1.0f / (1.0f + __expf(-oIn));
    const float cn2 = ft * c[idx] + it * ch;
    const float nn2 = ft * n[idx] + it;
    const float hn  = ot * (cn2 / (nn2 + 1e-8f));
    c[idx] = cn2; n[idx] = nn2; m[idx] = mn; h[idx] = hn;
    hbNew[idx] = f2bf(hn);
  }
}

// ---- Per-step output GEMM: out[:,t,:] = h_new @ Wout + bout -----------------
// 16 blocks x 256 thr (8 waves); TDM double-buffers the 2D Wout slab.
__global__ void __launch_bounds__(256)
lstm_out_wmma(const bf16* __restrict__ hb, const bf16* __restrict__ WoutT,
              const float* __restrict__ bout, float* __restrict__ out, int t) {
  extern __shared__ char smem[];
  bf16* ldsB = (bf16*)smem;                 // 2 buffers x 64 rows x BROW
  const size_t BUFE = (size_t)64 * BROW;
  const unsigned BUFBYTES = (unsigned)(BUFE * 2);      // 17,408 B

  const int tid  = threadIdx.x;
  const int lane = tid & 31;
  const int wave = tid >> 5;
  const int nBlock = blockIdx.x * 64;
  const int m0 = wave * 16;
  const unsigned ldsBase = __builtin_amdgcn_groupstaticsize();
  const bf16* wSlab = WoutT + (size_t)nBlock * Hsz;

  const int NST = Hsz / KC;                 // 8 stages
  if (wave == 0) {
    tdm_load_to_lds(wSlab, ldsBase, 32u, 64u, 0u,
                    (unsigned long long)(Hsz * 2 / 8), 0ull);
    __builtin_amdgcn_s_wait_tensorcnt(0);
  }
  __syncthreads();

  v8f a0 = {}, a1 = {}, a2 = {}, a3 = {};
  const int fn = lane & 15;
  const int fko = (lane >> 4) * 8;

  for (int s = 0; s < NST; ++s) {
    const bf16* cur = ldsB + (size_t)(s & 1) * BUFE;
    const bool more = (s + 1 < NST);
    if (more && wave == 0) {
      tdm_load_to_lds(wSlab + (size_t)(s + 1) * KC,
                      ldsBase + (unsigned)((s + 1) & 1) * BUFBYTES,
                      32u, 64u, 0u,
                      (unsigned long long)(Hsz * 2 / 8), 0ull);
    }
#pragma unroll
    for (int kk = 0; kk < KC; kk += 32) {
      v16bf A = load_A_h(hb, m0, s * KC + kk, lane);
      const bf16* fb = cur + (size_t)fn * BROW + kk + fko;
      v16bf B0 = lds_fragB(fb);
      v16bf B1 = lds_fragB(fb + 16 * BROW);
      v16bf B2 = lds_fragB(fb + 32 * BROW);
      v16bf B3 = lds_fragB(fb + 48 * BROW);
      WMMA_BF16(a0, A, B0);
      WMMA_BF16(a1, A, B1);
      WMMA_BF16(a2, A, B2);
      WMMA_BF16(a3, A, B3);
    }
    if (more) {
      if (wave == 0) __builtin_amdgcn_s_wait_tensorcnt(0);
      __syncthreads();
    }
  }

  const int half = lane >> 4;
#pragma unroll
  for (int r = 0; r < 8; ++r) {
    const int row = m0 + r + 8 * half;
    float* o = out + ((size_t)row * Ssz + t) * Osz + nBlock + fn;
    o[ 0] = a0[r] + bout[nBlock + fn +  0];
    o[16] = a1[r] + bout[nBlock + fn + 16];
    o[32] = a2[r] + bout[nBlock + fn + 32];
    o[48] = a3[r] + bout[nBlock + fn + 48];
  }
}

// ---- Setup kernels ----------------------------------------------------------
__global__ void cvt_bf16(const float* __restrict__ src, bf16* __restrict__ dst, size_t nElem) {
  const size_t stride = (size_t)gridDim.x * blockDim.x;
  for (size_t i = (size_t)blockIdx.x * blockDim.x + threadIdx.x; i < nElem; i += stride)
    dst[i] = f2bf(src[i]);
}

// WT[n][k] = (bf16) W[k][n]   (W is rows x cols = K x N row-major)
__global__ void cvt_wT(const float* __restrict__ W, bf16* __restrict__ WT,
                       int rows, int cols) {
  const size_t nElem  = (size_t)rows * cols;
  const size_t stride = (size_t)gridDim.x * blockDim.x;
  for (size_t i = (size_t)blockIdx.x * blockDim.x + threadIdx.x; i < nElem; i += stride) {
    const int nIdx = (int)(i / rows);
    const int k    = (int)(i % rows);
    WT[i] = f2bf(W[(size_t)k * cols + nIdx]);
  }
}

__global__ void init_state(float* __restrict__ states, bf16* __restrict__ hb0) {
  const int i = blockIdx.x * blockDim.x + threadIdx.x;
  if (i < 4 * BH) states[i] = 0.0f;
  if (i < BH)     hb0[i] = f2bf(0.0f);
}

// ---- Host launch ------------------------------------------------------------
extern "C" void kernel_launch(void* const* d_in, const int* in_sizes, int n_in,
                              void* d_out, int out_size, void* d_ws, size_t ws_size,
                              hipStream_t stream) {
  (void)in_sizes; (void)n_in; (void)out_size; (void)ws_size;
  const float* x    = (const float*)d_in[0];
  const float* Wf   = (const float*)d_in[1];
  const float* bf_  = (const float*)d_in[2];
  const float* Wi   = (const float*)d_in[3];
  const float* bi_  = (const float*)d_in[4];
  const float* Wc   = (const float*)d_in[5];
  const float* bc_  = (const float*)d_in[6];
  const float* Wo   = (const float*)d_in[7];
  const float* bo_  = (const float*)d_in[8];
  const float* Wout = (const float*)d_in[9];
  const float* bout = (const float*)d_in[10];

  // workspace layout (256B-aligned partitions): ~146.5 MB total
  char* ws = (char*)d_ws;
  bf16* xb    = (bf16*)(ws);                                    // B*S*I bf16 = 128 MB
  bf16* WT    = (bf16*)(ws + (size_t)Bsz * Ssz * Isz * 2);      // 4*H*KK bf16 = 16 MB
  bf16* WoutT = (bf16*)((char*)WT + (size_t)4 * Hsz * KK * 2);  // O*H bf16 = 2 MB
  bf16* hb0   = (bf16*)((char*)WoutT + (size_t)Osz * Hsz * 2);  // B*H bf16 (ping)
  bf16* hb1   = hb0 + (size_t)BH;                               // B*H bf16 (pong)

  float* out = (float*)d_out;                       // [B][S][O]
  float* hS  = out + (size_t)Bsz * Ssz * Osz;       // final states in d_out tail
  float* cS  = hS + BH;
  float* nS  = cS + BH;
  float* mS  = nS + BH;

  const size_t SMEM_FUSED = (size_t)2 * 4 * 64 * BROW * sizeof(bf16); // 139,264 B
  const size_t SMEM_OUT   = (size_t)2 * 64 * BROW * sizeof(bf16);     //  34,816 B

  // Phase 0: stage bf16 copies (x, transposed weights), zero states + hb0.
  cvt_bf16<<<4096, 256, 0, stream>>>(x, xb, (size_t)Bsz * Ssz * Isz);
  cvt_wT<<<2048, 256, 0, stream>>>(Wf,   WT + (size_t)0 * Hsz * KK, KK, Hsz);
  cvt_wT<<<2048, 256, 0, stream>>>(Wi,   WT + (size_t)1 * Hsz * KK, KK, Hsz);
  cvt_wT<<<2048, 256, 0, stream>>>(Wc,   WT + (size_t)2 * Hsz * KK, KK, Hsz);
  cvt_wT<<<2048, 256, 0, stream>>>(Wo,   WT + (size_t)3 * Hsz * KK, KK, Hsz);
  cvt_wT<<<1024, 256, 0, stream>>>(Wout, WoutT, Hsz, Osz);
  init_state<<<(4 * BH + 255) / 256, 256, 0, stream>>>(hS, hb0);

  // Phase 1: sequential scan, 2 launches per step. Weights stay hot in L2;
  // h ping-pongs between hb0/hb1 (fused kernel reads prev, writes next).
  for (int t = 0; t < Ssz; ++t) {
    const bf16* hPrev = (t & 1) ? hb1 : hb0;
    bf16*       hNext = (t & 1) ? hb0 : hb1;
    lstm_step_fused<<<16, 1024, SMEM_FUSED, stream>>>(
        xb, hPrev, WT, bf_, bi_, bc_, bo_, hS, cS, nS, mS, hNext, t);
    lstm_out_wmma<<<16, 256, SMEM_OUT, stream>>>(hNext, WoutT, bout, out, t);
  }
}
